// Attention_45174466020040
// MI455X (gfx1250) — matrix operand order repaired and checked by reading, not compiled, not run on hardware
//
#include <hip/hip_runtime.h>
#include <hip/hip_bf16.h>
#include <math.h>

#define HIDDEN 4096
#define NH 16
#define HD 256            // head dim
#define ROT 128           // rotary dim
#define BB 4
#define SS 1024

typedef __attribute__((ext_vector_type(16))) __bf16 v16bf;
typedef __attribute__((ext_vector_type(8)))  __bf16 v8bf;
typedef __attribute__((ext_vector_type(8)))  float  v8f;
typedef int v4i __attribute__((vector_size(16)));   // matches async-builtin pointee

union BF16x16 { v16bf v; v8bf h[2]; };

#if defined(__HIP_DEVICE_COMPILE__) && __has_builtin(__builtin_amdgcn_global_load_async_to_lds_b128)
#define HAVE_ASYNC 1
typedef __attribute__((address_space(1))) v4i* gptr_v4i;
typedef __attribute__((address_space(3))) v4i* lptr_v4i;
#else
#define HAVE_ASYNC 0
#endif

__device__ __forceinline__ v8f wmma_bf16(const BF16x16& a, const BF16x16& b, v8f c) {
    return __builtin_amdgcn_wmma_f32_16x16x32_bf16(
        false, a.v, false, b.v, (short)0, c, false, false);
}

__device__ __forceinline__ unsigned pack_bf16(float a, float b) {
    union { __bf16 h[2]; unsigned u; } p;
    p.h[0] = (__bf16)a; p.h[1] = (__bf16)b;
    return p.u;
}

// ---------------------------------------------------------------------------
// C = A(f32) @ B(f32), bf16 WMMA, f32 accumulate.
// Block tile 128x256, 256 threads = 8 waves (2x4), wave tile 64x64, K-step 32,
// double-buffered LDS (48KB) + prefetch 2 tiles ahead.
// ---------------------------------------------------------------------------
template <typename TC>
__global__ __launch_bounds__(256)
void gemm_wmma_kernel(const float* __restrict__ A, const float* __restrict__ B,
                      TC* __restrict__ C, int M, int N, int K)
{
    __shared__ __bf16 As[2][128 * 32];   // [m][k]
    __shared__ __bf16 Bs[2][256 * 32];   // [n][k]  (transposed on store)

    const int tid  = threadIdx.x;
    const int lane = tid & 31;
    const int wave = tid >> 5;
    const int wm   = wave >> 2;       // 0..1 -> 64 rows each
    const int wn   = wave & 3;        // 0..3 -> 64 cols each
    const int hi   = lane >> 4;
    const int ln   = lane & 15;
    const int m0   = blockIdx.y * 128;
    const int n0   = blockIdx.x * 256;

    const v8f vzero = {0.f,0.f,0.f,0.f,0.f,0.f,0.f,0.f};
    v8f acc[4][4];
    #pragma unroll
    for (int i = 0; i < 4; ++i)
        #pragma unroll
        for (int j = 0; j < 4; ++j) acc[i][j] = vzero;

    const int arow = tid >> 1;            // 0..127
    const int ak0  = (tid & 1) * 16;      // 0/16
    const int bk   = tid >> 3;            // 0..31
    const int bn0  = (tid & 7) * 32;      // 0..224

    auto stage = [&](int k0, int buf) {
        // A tile: 16 consecutive k per thread, packed 2xbf16 ds_store_b32
        const float* asrc = A + (size_t)(m0 + arow) * K + k0 + ak0;
        float4 av[4];
        #pragma unroll
        for (int i = 0; i < 4; ++i) av[i] = ((const float4*)asrc)[i];
        unsigned* ad = (unsigned*)&As[buf][arow * 32 + ak0];
        #pragma unroll
        for (int i = 0; i < 4; ++i) {
            ad[2*i+0] = pack_bf16(av[i].x, av[i].y);
            ad[2*i+1] = pack_bf16(av[i].z, av[i].w);
        }
        // B tile: 32 consecutive n per thread, transposed store Bs[n][k]
        const float* bsrc = B + (size_t)(k0 + bk) * N + n0 + bn0;
        float4 bv[8];
        #pragma unroll
        for (int i = 0; i < 8; ++i) bv[i] = ((const float4*)bsrc)[i];
        #pragma unroll
        for (int i = 0; i < 8; ++i) {
            Bs[buf][(bn0 + 4*i + 0) * 32 + bk] = (__bf16)bv[i].x;
            Bs[buf][(bn0 + 4*i + 1) * 32 + bk] = (__bf16)bv[i].y;
            Bs[buf][(bn0 + 4*i + 2) * 32 + bk] = (__bf16)bv[i].z;
            Bs[buf][(bn0 + 4*i + 3) * 32 + bk] = (__bf16)bv[i].w;
        }
    };

    const int nk = K / 32;
    stage(0, 0);

    const int akb = hi * 8;    // A lane layout: lo lanes K {0..7,16..23}, hi {8..15,24..31}
    const int bkb = hi * 8;    // B lane layout: lo lanes K {0..7,16..23}, hi {8..15,24..31}

    for (int kt = 0; kt < nk; ++kt) {
        const int buf = kt & 1;
        __syncthreads();
        if (kt + 1 < nk) stage((kt + 1) * 32, buf ^ 1);
        if (kt + 2 < nk) {
            __builtin_prefetch(A + (size_t)(m0 + arow) * K + (kt + 2) * 32 + ak0, 0, 1);
            __builtin_prefetch(B + (size_t)((kt + 2) * 32 + bk) * N + n0 + bn0, 0, 1);
        }

        BF16x16 afr[4];
        #pragma unroll
        for (int mt = 0; mt < 4; ++mt) {
            int m = wm * 64 + mt * 16 + ln;
            afr[mt].h[0] = *(const v8bf*)&As[buf][m * 32 + akb];
            afr[mt].h[1] = *(const v8bf*)&As[buf][m * 32 + 16 + akb];
        }
        #pragma unroll
        for (int nt = 0; nt < 4; ++nt) {
            int n = wn * 64 + nt * 16 + ln;
            BF16x16 bfr;
            bfr.h[0] = *(const v8bf*)&Bs[buf][n * 32 + bkb];
            bfr.h[1] = *(const v8bf*)&Bs[buf][n * 32 + bkb + 16];
            #pragma unroll
            for (int mt = 0; mt < 4; ++mt)
                acc[mt][nt] = wmma_bf16(afr[mt], bfr, acc[mt][nt]);
        }
    }

    // Epilogue: C layout = VGPR r -> row r + 8*hi, col = ln
    #pragma unroll
    for (int mt = 0; mt < 4; ++mt)
        #pragma unroll
        for (int nt = 0; nt < 4; ++nt) {
            int col = n0 + wn * 64 + nt * 16 + ln;
            #pragma unroll
            for (int r = 0; r < 8; ++r) {
                int row = m0 + wm * 64 + mt * 16 + r + 8 * hi;
                C[(size_t)row * N + col] = (TC)acc[mt][nt][r];
            }
        }
}

// ---------------------------------------------------------------------------
// RoPE + repack: qkv bf16 [tok][3*HIDDEN] -> q/k (rotated), v copied,
// each as bf16 [b][h][s][HD].
// ---------------------------------------------------------------------------
__global__ __launch_bounds__(256)
void rope_repack_kernel(const __bf16* __restrict__ qkv,
                        const int* __restrict__ pids,
                        __bf16* __restrict__ qo, __bf16* __restrict__ ko,
                        __bf16* __restrict__ vo)
{
    size_t idx = (size_t)blockIdx.x * blockDim.x + threadIdx.x;
    const size_t per = (size_t)BB * SS * NH * HD;
    if (idx >= 3 * per) return;
    int comp = (int)(idx / per);      // 0=q 1=k 2=v
    size_t r = idx % per;
    int d = (int)(r % HD);
    size_t t = r / HD;
    int h = (int)(t % NH);
    size_t tok = t / NH;              // b*S + s
    int s = (int)(tok % SS);
    int b = (int)(tok / SS);

    const __bf16* src = qkv + tok * (size_t)(3 * HIDDEN) + comp * HIDDEN + h * HD;
    float out;
    if (comp == 2 || d >= ROT) {
        out = (float)src[d];
    } else {
        const int half = ROT / 2;           // 64
        int f = (d < half) ? d : d - half;
        float inv = __expf(-((float)f / 64.0f) * 9.210340371976184f);
        float ang = (float)pids[tok] * inv;
        float c = __cosf(ang), sn = __sinf(ang);
        float x1 = (float)src[f], x2 = (float)src[f + half];
        out = (d < half) ? (x1 * c - x2 * sn) : (x2 * c + x1 * sn);
    }
    size_t dst = (((size_t)b * NH + h) * SS + s) * HD + d;
    __bf16 ob = (__bf16)out;
    if (comp == 0) qo[dst] = ob;
    else if (comp == 1) ko[dst] = ob;
    else vo[dst] = ob;
}

// ---------------------------------------------------------------------------
// Causal flash attention. Block = 128 threads (4 waves), 64 query rows per
// block (16 per wave), HD=256 accumulated in f32 (16 x v8f per wave).
// K-loop steps 32 keys. K tile async-copied to LDS (if available),
// V tile staged transposed.
// ---------------------------------------------------------------------------
__global__ __launch_bounds__(128)
void attn_flash_kernel(const __bf16* __restrict__ Q, const __bf16* __restrict__ Kg,
                       const __bf16* __restrict__ V, float* __restrict__ O)
{
    __shared__ __bf16 Ks[32 * HD];       // [key][d]   16 KB
    __shared__ __bf16 Vs[HD * 32];       // [d][key]   16 KB
    __shared__ __bf16 Ps[4][16 * 32];    // per-wave P 4 KB

    const int tid  = threadIdx.x;
    const int lane = tid & 31;
    const int wave = tid >> 5;
    const int hi   = lane >> 4;
    const int ln   = lane & 15;
    const int bh   = blockIdx.y;         // b*NH + h
    const int b    = bh >> 4, h = bh & 15;
    const int qb   = blockIdx.x;         // 0..15
    const int qbase = qb * 64 + wave * 16;

    const __bf16* Qp = Q + (size_t)bh * SS * HD;
    const __bf16* Kp = Kg + (size_t)bh * SS * HD;
    const __bf16* Vp = V + (size_t)bh * SS * HD;

    // Q fragments (A layout), held for the whole K loop: 8 chunks of K=32
    BF16x16 qf[8];
    {
        const __bf16* qr = Qp + (size_t)(qbase + ln) * HD;
        const int kb = hi * 8;
        #pragma unroll
        for (int c = 0; c < 8; ++c) {
            qf[c].h[0] = *(const v8bf*)(qr + c * 32 + kb);
            qf[c].h[1] = *(const v8bf*)(qr + c * 32 + 16 + kb);
        }
    }

    const v8f vzero = {0.f,0.f,0.f,0.f,0.f,0.f,0.f,0.f};
    v8f acc[16];
    #pragma unroll
    for (int t = 0; t < 16; ++t) acc[t] = vzero;
    float rowm[8], rowl[8];
    #pragma unroll
    for (int r = 0; r < 8; ++r) { rowm[r] = -3.0e38f; rowl[r] = 0.f; }

    const float scale = 0.0625f;         // HD^-0.5
    const int nkt = 2 * qb + 2;          // causal: key blocks needed

    for (int kt = 0; kt < nkt; ++kt) {
        __syncthreads();
#if HAVE_ASYNC
        {   // K tile: contiguous 16KB via async global->LDS (ASYNCcnt path)
            const __bf16* gsrc = Kp + (size_t)kt * 32 * HD;
            #pragma unroll
            for (int i = 0; i < 8; ++i) {
                int off = (tid + i * 128) * 8;   // elements; 16B chunks
                __builtin_amdgcn_global_load_async_to_lds_b128(
                    (gptr_v4i)(gsrc + off), (lptr_v4i)(Ks + off), 0, 0);
            }
        }
#else
        {   // K tile: contiguous 16KB copy
            const uint4* src = (const uint4*)(Kp + (size_t)kt * 32 * HD);
            uint4* dst = (uint4*)Ks;
            #pragma unroll
            for (int i = 0; i < 8; ++i) dst[tid + i * 128] = src[tid + i * 128];
        }
#endif
        {   // V tile transposed: Vs[d][key]
            int key = tid >> 2;
            int d0  = (tid & 3) * 64;
            const __bf16* src = Vp + (size_t)(kt * 32 + key) * HD + d0;
            #pragma unroll
            for (int i = 0; i < 64; ++i)
                Vs[(d0 + i) * 32 + key] = src[i];
        }
#if HAVE_ASYNC
        asm volatile("s_wait_asynccnt 0x0" ::: "memory");
#endif
        __syncthreads();

        // scores: two 16-key subtiles, K-dim = HD over 8 WMMA steps
        v8f sc[2];
        #pragma unroll
        for (int ks = 0; ks < 2; ++ks) {
            sc[ks] = vzero;
            int key = ks * 16 + ln;
            int kb  = hi * 8;
            #pragma unroll
            for (int c = 0; c < 8; ++c) {
                BF16x16 kf;
                kf.h[0] = *(const v8bf*)&Ks[key * HD + c * 32 + kb];
                kf.h[1] = *(const v8bf*)&Ks[key * HD + c * 32 + kb + 16];
                sc[ks] = wmma_bf16(qf[c], kf, sc[ks]);
            }
        }

        // online softmax (16-lane-half butterflies match C layout)
        const int key0 = kt * 32 + ln, key1 = key0 + 16;
        #pragma unroll
        for (int r = 0; r < 8; ++r) {
            int row = qbase + r + 8 * hi;
            float s0 = sc[0][r] * scale;
            float s1 = sc[1][r] * scale;
            if (key0 > row) s0 = -3.0e38f;
            if (key1 > row) s1 = -3.0e38f;
            float mx = fmaxf(s0, s1);
            mx = fmaxf(mx, __shfl_xor(mx, 1));
            mx = fmaxf(mx, __shfl_xor(mx, 2));
            mx = fmaxf(mx, __shfl_xor(mx, 4));
            mx = fmaxf(mx, __shfl_xor(mx, 8));
            float mnew = fmaxf(rowm[r], mx);
            float corr = __expf(rowm[r] - mnew);
            float p0 = __expf(s0 - mnew);
            float p1 = __expf(s1 - mnew);
            float ps = p0 + p1;
            ps += __shfl_xor(ps, 1);
            ps += __shfl_xor(ps, 2);
            ps += __shfl_xor(ps, 4);
            ps += __shfl_xor(ps, 8);
            rowl[r] = rowl[r] * corr + ps;
            rowm[r] = mnew;
            #pragma unroll
            for (int t = 0; t < 16; ++t) acc[t][r] *= corr;
            Ps[wave][(r + 8 * hi) * 32 + ln]      = (__bf16)p0;
            Ps[wave][(r + 8 * hi) * 32 + 16 + ln] = (__bf16)p1;
        }
        asm volatile("s_wait_dscnt 0x0" ::: "memory");

        // PV: A = P (16x32), B = V^T slices (32x16 per d-tile)
        BF16x16 pf;
        {
            const int kb = hi * 8;
            pf.h[0] = *(const v8bf*)&Ps[wave][ln * 32 + kb];
            pf.h[1] = *(const v8bf*)&Ps[wave][ln * 32 + 16 + kb];
        }
        const int vkb = hi * 8;
        #pragma unroll
        for (int t = 0; t < 16; ++t) {
            BF16x16 vf;
            vf.h[0] = *(const v8bf*)&Vs[(t * 16 + ln) * 32 + vkb];
            vf.h[1] = *(const v8bf*)&Vs[(t * 16 + ln) * 32 + vkb + 16];
            acc[t] = wmma_bf16(pf, vf, acc[t]);
        }
    }

    // normalize + write attn output f32 [b][s][h*HD + d]
    #pragma unroll
    for (int t = 0; t < 16; ++t)
        #pragma unroll
        for (int r = 0; r < 8; ++r) {
            int row = qbase + r + 8 * hi;
            float o = acc[t][r] / rowl[r];
            size_t tok = (size_t)b * SS + row;
            O[tok * HIDDEN + h * HD + t * 16 + ln] = o;
        }
}

// ---------------------------------------------------------------------------
// Launch: qkv-GEMM -> RoPE/repack -> flash attention -> out-GEMM
// ws layout: [qkv bf16 100.7MB | q bf16 33.6 | k bf16 33.6 | v bf16 33.6]
// attn f32 (67MB) reuses the (dead) qkv region.
// ---------------------------------------------------------------------------
extern "C" void kernel_launch(void* const* d_in, const int* in_sizes, int n_in,
                              void* d_out, int out_size, void* d_ws, size_t ws_size,
                              hipStream_t stream) {
    const int*   pids   = (const int*)d_in[0];
    const float* hidden = (const float*)d_in[1];
    const float* w_qkv  = (const float*)d_in[2];
    const float* w_out  = (const float*)d_in[3];

    const size_t SZ_QKV = (size_t)BB * SS * 3 * HIDDEN * sizeof(__bf16);
    const size_t SZ_QH  = (size_t)BB * SS * HIDDEN * sizeof(__bf16);

    char* ws = (char*)d_ws;
    __bf16* qkv = (__bf16*)ws;
    __bf16* qo  = (__bf16*)(ws + SZ_QKV);
    __bf16* ko  = (__bf16*)(ws + SZ_QKV + SZ_QH);
    __bf16* vo  = (__bf16*)(ws + SZ_QKV + 2 * SZ_QH);
    float*  attn = (float*)ws;   // reuses qkv region after RoPE consumes it

    const int M = BB * SS;                  // 4096 tokens

    // 1) qkv = hidden @ w_qkv   (4096 x 12288 x 4096)
    gemm_wmma_kernel<__bf16><<<dim3(3 * HIDDEN / 256, M / 128), 256, 0, stream>>>(
        hidden, w_qkv, qkv, M, 3 * HIDDEN, HIDDEN);

    // 2) RoPE + repack to [b][h][s][d] bf16
    {
        size_t total = 3ull * BB * SS * NH * HD;
        int blocks = (int)((total + 255) / 256);
        rope_repack_kernel<<<blocks, 256, 0, stream>>>(qkv, pids, qo, ko, vo);
    }

    // 3) causal flash attention -> attn f32 [tok][HIDDEN]
    attn_flash_kernel<<<dim3(SS / 64, BB * NH), 128, 0, stream>>>(qo, ko, vo, attn);

    // 4) out = attn @ w_out  (4096 x 4096 x 4096)
    gemm_wmma_kernel<float><<<dim3(HIDDEN / 256, M / 128), 256, 0, stream>>>(
        attn, w_out, (float*)d_out, M, HIDDEN, HIDDEN);
}
